// SparseAttention_7902739825181
// MI455X (gfx1250) — matrix operand order repaired and checked by reading, not compiled, not run on hardware
//
#include <hip/hip_runtime.h>
#include <hip/hip_bf16.h>

typedef __attribute__((ext_vector_type(16))) _Float16 v16h;
typedef __attribute__((ext_vector_type(8)))  _Float16 v8h;
typedef __attribute__((ext_vector_type(8)))  float    v8f;

// Problem constants (fixed by setup_inputs): B=2,H=16,L=2048,D=128
#define BH_N   32      // B*H
#define LSEQ   2048
#define DDIM   128
#define NBLK   32      // nQ == nK
#define BLK    64
#define TOPK   8       // 0.25 * 32

// LDS strides (halves) — padded to break bank conflicts; multiples of 8 keep
// ds_load_b128 fragment loads 16B-aligned.
#define LDQ    136     // Qs / Ks row stride
#define LDV    72      // Vt / P row stride

// ---------------- WMMA fragment loaders (wave32) ----------------
// A-matrix 16x32 f16 (ISA 7.12.2): lane half sel = lane>>4, row = lane&15.
// half h -> K = h + 8*sel + 8*(h>=8)  => two contiguous 8-half (16B) loads.
__device__ __forceinline__ v16h load_a_frag(const _Float16* __restrict__ base, int ld) {
    const int lane = threadIdx.x & 31;
    const int r    = lane & 15;
    const int sel  = lane >> 4;
    const _Float16* p = base + r * ld + sel * 8;
    v8h lo = *(const v8h*)(p);       // K = sel*8 .. sel*8+7
    v8h hi = *(const v8h*)(p + 16);  // K = 16+sel*8 .. 16+sel*8+7
    v16h o;
#pragma unroll
    for (int i = 0; i < 8; ++i) { o[i] = lo[i]; o[i + 8] = hi[i]; }
    return o;
}

// B-matrix 32x16 f16: same K order as A: half h -> K = h + 8*sel + 8*(h>=8);
// col n = lane&15. Two contiguous 8-half (16B) loads.
__device__ __forceinline__ v16h load_b_frag(const _Float16* __restrict__ base, int ld) {
    const int lane = threadIdx.x & 31;
    const int n    = lane & 15;
    const int sel  = lane >> 4;
    const _Float16* p = base + n * ld + sel * 8;
    v8h lo = *(const v8h*)(p);       // K = sel*8 .. sel*8+7
    v8h hi = *(const v8h*)(p + 16);  // K = 16+sel*8 .. 16+sel*8+7
    v16h o;
#pragma unroll
    for (int i = 0; i < 8; ++i) { o[i] = lo[i]; o[i + 8] = hi[i]; }
    return o;
}

__device__ __forceinline__ v8f wmma_f16(v16h a, v16h b, v8f c) {
    return __builtin_amdgcn_wmma_f32_16x16x32_f16(
        /*neg_a=*/false, a, /*neg_b=*/false, b,
        /*c_mod=*/(short)0, c, /*reuse_a=*/false, /*reuse_b=*/false);
}

// ---------------- Phase 1: block map (mean pool + scores + top-8) ----------------
__global__ __launch_bounds__(256) void sa_blockmap(const float* __restrict__ q,
                                                   const float* __restrict__ k,
                                                   int* __restrict__ lut) {
    const int bh  = blockIdx.x;      // 0..31
    const int tid = threadIdx.x;
    __shared__ float qp[NBLK][DDIM];
    __shared__ float kp[NBLK][DDIM];
    __shared__ float sc[NBLK][NBLK];

    const size_t gbase = (size_t)bh * LSEQ * DDIM;
    const float inv64 = 1.0f / 64.0f;

    for (int i = tid; i < NBLK * DDIM; i += 256) {
        const int blk = i >> 7, d = i & (DDIM - 1);
        const float* qq = q + gbase + (size_t)blk * BLK * DDIM + d;
        const float* kk = k + gbase + (size_t)blk * BLK * DDIM + d;
        float sq = 0.f, sk = 0.f;
        for (int r = 0; r < BLK; ++r) { sq += qq[r * DDIM]; sk += kk[r * DDIM]; }
        qp[blk][d] = sq * inv64;
        kp[blk][d] = sk * inv64;
    }
    __syncthreads();

    for (int i = tid; i < NBLK * NBLK; i += 256) {
        const int qb = i >> 5, kb = i & 31;
        float s = 0.f;
        for (int d = 0; d < DDIM; ++d) s += qp[qb][d] * kp[kb][d];
        sc[qb][kb] = s;
    }
    __syncthreads();

    if (tid < NBLK) {  // one thread per query block row: top-8, ties -> lowest index
        const int row = tid;
        for (int t = 0; t < TOPK; ++t) {
            float best = -__builtin_inff();
            int bi = 0;
            for (int j = 0; j < NBLK; ++j) {
                const float v = sc[row][j];
                if (v > best) { best = v; bi = j; }
            }
            sc[row][bi] = -__builtin_inff();
            lut[(bh * NBLK + row) * TOPK + t] = bi;
        }
    }
}

// ---------------- Phase 2: block-sparse flash attention ----------------
// Grid: B*H*nQ = 1024 workgroups of 128 threads (4 waves).
// Each wave owns a 16-row Q strip; loops over 8 selected key blocks with
// online softmax; O accumulated in 8 f32 WMMA accumulators per lane.
__global__ __launch_bounds__(128) void sa_attn(const float* __restrict__ q,
                                               const float* __restrict__ k,
                                               const float* __restrict__ v,
                                               const int* __restrict__ lut,
                                               float* __restrict__ out) {
    const int wg   = blockIdx.x;      // bh*32 + qb
    const int bh   = wg >> 5;
    const int qb   = wg & 31;
    const int tid  = threadIdx.x;
    const int wave = tid >> 5;
    const int lane = tid & 31;
    const int ln   = lane & 15;
    const int sel  = lane >> 4;

    __shared__ __align__(16) _Float16 Qs[BLK * LDQ];        // 64 x 128 (pad 136)
    __shared__ __align__(16) _Float16 Ks[BLK * LDQ];        // 64 x 128 (pad 136)
    __shared__ __align__(16) _Float16 Vt[DDIM * LDV];       // 128 x 64 transposed (pad 72)
    __shared__ __align__(16) _Float16 Pl[4 * 16 * LDV];     // per-wave 16 x 64 P strip

    const size_t gbase = (size_t)bh * LSEQ * DDIM;
    const float* qg = q + gbase + (size_t)qb * BLK * DDIM;

    for (int i = tid; i < BLK * DDIM; i += 128) {
        const int r = i >> 7, c = i & (DDIM - 1);
        Qs[r * LDQ + c] = (_Float16)qg[i];
    }

    v8f zero;
#pragma unroll
    for (int i = 0; i < 8; ++i) zero[i] = 0.f;

    v8f oacc[8];
#pragma unroll
    for (int t = 0; t < 8; ++t) oacc[t] = zero;
    float m[8], l[8];
#pragma unroll
    for (int r = 0; r < 8; ++r) { m[r] = -__builtin_inff(); l[r] = 0.f; }

    const float scale = 0.08838834764831845f;  // 1/sqrt(128)
    const int* mylut = lut + wg * TOPK;

    for (int jb = 0; jb < TOPK; ++jb) {
        __syncthreads();  // previous iteration's consumers done with Ks/Vt
        const int kblk = mylut[jb];
        const float* kg = k + gbase + (size_t)kblk * BLK * DDIM;
        const float* vg = v + gbase + (size_t)kblk * BLK * DDIM;
        for (int i = tid; i < BLK * DDIM; i += 128) {
            const int r = i >> 7, c = i & (DDIM - 1);
            Ks[r * LDQ + c] = (_Float16)kg[i];
            Vt[c * LDV + r] = (_Float16)vg[i];   // transpose V into LDS
        }
        if (jb + 1 < TOPK) {  // prefetch next selected block (global_prefetch_b8)
            const int nb = mylut[jb + 1];
            __builtin_prefetch(k + gbase + (size_t)nb * BLK * DDIM + tid * 64, 0, 0);
            __builtin_prefetch(v + gbase + (size_t)nb * BLK * DDIM + tid * 64, 0, 0);
        }
        __syncthreads();

        // ---- S = Q * K^T : this wave's 16x64 strip, 4 n-tiles, D in 4 steps of 32
        v8f sacc[4];
#pragma unroll
        for (int t = 0; t < 4; ++t) sacc[t] = zero;
#pragma unroll
        for (int ks = 0; ks < 4; ++ks) {
            v16h a = load_a_frag(&Qs[(wave * 16) * LDQ + ks * 32], LDQ);
#pragma unroll
            for (int t = 0; t < 4; ++t) {
                v16h b = load_b_frag(&Ks[(t * 16) * LDQ + ks * 32], LDQ);
                sacc[t] = wmma_f16(a, b, sacc[t]);
            }
        }

        // ---- online softmax (rows live across 16-lane halves; xor-shuffles <16)
        float pj[4][8];
#pragma unroll
        for (int r = 0; r < 8; ++r) {
            float mx = fmaxf(fmaxf(sacc[0][r], sacc[1][r]),
                             fmaxf(sacc[2][r], sacc[3][r])) * scale;
#pragma unroll
            for (int off = 1; off < 16; off <<= 1)
                mx = fmaxf(mx, __shfl_xor(mx, off, 32));
            const float mn = fmaxf(m[r], mx);
            float rs = 0.f;
#pragma unroll
            for (int t = 0; t < 4; ++t) {
                const float p = __expf(sacc[t][r] * scale - mn);
                pj[t][r] = p;
                rs += p;
            }
#pragma unroll
            for (int off = 1; off < 16; off <<= 1)
                rs += __shfl_xor(rs, off, 32);
            const float alpha = __expf(m[r] - mn);
            l[r] = l[r] * alpha + rs;
            m[r] = mn;
#pragma unroll
            for (int t = 0; t < 8; ++t) oacc[t][r] *= alpha;
        }

        // ---- stash P strip (wave-private LDS; same-wave DS ops stay ordered)
        _Float16* pw = &Pl[wave * 16 * LDV];
#pragma unroll
        for (int t = 0; t < 4; ++t)
#pragma unroll
            for (int r = 0; r < 8; ++r)
                pw[(r + sel * 8) * LDV + t * 16 + ln] = (_Float16)pj[t][r];

        // ---- O += P * V : 8 d-tiles, 64 keys in 2 steps of 32
#pragma unroll
        for (int ks = 0; ks < 2; ++ks) {
            v16h a = load_a_frag(&pw[ks * 32], LDV);
#pragma unroll
            for (int t = 0; t < 8; ++t) {
                v16h b = load_b_frag(&Vt[(t * 16) * LDV + ks * 32], LDV);
                oacc[t] = wmma_f16(a, b, oacc[t]);
            }
        }
    }

    // ---- normalize and write out (C layout: row = r + 8*sel, col = ln)
    float invl[8];
#pragma unroll
    for (int r = 0; r < 8; ++r) invl[r] = 1.0f / l[r];
#pragma unroll
    for (int t = 0; t < 8; ++t) {
#pragma unroll
        for (int r = 0; r < 8; ++r) {
            const int row = qb * BLK + wave * 16 + r + sel * 8;
            const int col = t * 16 + ln;
            out[gbase + (size_t)row * DDIM + col] = oacc[t][r] * invl[r];
        }
    }
}

// ---------------- launcher ----------------
extern "C" void kernel_launch(void* const* d_in, const int* in_sizes, int n_in,
                              void* d_out, int out_size, void* d_ws, size_t ws_size,
                              hipStream_t stream) {
    const float* q = (const float*)d_in[0];
    const float* k = (const float*)d_in[1];
    const float* v = (const float*)d_in[2];
    float* out = (float*)d_out;
    int* lut = (int*)d_ws;  // B*H*nQ*TOPK = 8192 ints

    sa_blockmap<<<dim3(BH_N), dim3(256), 0, stream>>>(q, k, lut);
    sa_attn<<<dim3(BH_N * NBLK), dim3(128), 0, stream>>>(q, k, v, lut, out);
}
